// QuantGruLayer_8710193676875
// MI455X (gfx1250) — matrix operand order repaired and checked by reading, not compiled, not run on hardware
//
#include <hip/hip_runtime.h>
#include <hip/hip_bf16.h>

typedef __attribute__((ext_vector_type(16))) __bf16 v16bf;
typedef __attribute__((ext_vector_type(8)))  float  v8f;

#define GB 64     // batch
#define GT 512    // time steps
#define GI 512    // input dim
#define GH 512    // hidden dim
#define NBLK 32   // persistent blocks (each owns 16 hidden columns)
#define NTHR 128  // 4 waves of 32

union BFrag { v16bf v; uint4 u[2]; };

__device__ inline v8f wmma_bf16(v16bf a, v16bf b, v8f c) {
  // D = A(16x32) * B(32x16) + C, f32 accumulate
  return __builtin_amdgcn_wmma_f32_16x16x32_bf16(false, a, false, b, (short)0, c,
                                                 false, false);
}

// B-fragment: lane holds column N=lane%16; K = {kb0..kb0+7, kb0+16..kb0+23}
__device__ inline v16bf ldfrag_b(const __bf16* p) {
  BFrag f;
  f.u[0] = *(const uint4*)(p);
  f.u[1] = *(const uint4*)(p + 16);
  return f.v;
}

// A-fragment (bf16 row-major source): lane row M=lane%16; K = {k0..k0+7, k0+16..k0+23}
__device__ inline v16bf ldfrag_a(const __bf16* row, int k0) {
  BFrag f;
  f.u[0] = *(const uint4*)(row + k0);
  f.u[1] = *(const uint4*)(row + k0 + 16);
  return f.v;
}

// A-fragment from fp32 row-major source, converted on the fly to bf16 (fallback path)
__device__ inline v16bf ldfrag_a_f32(const float* row, int k0) {
  v16bf v;
  float4 f0 = *(const float4*)(row + k0);
  float4 f1 = *(const float4*)(row + k0 + 4);
  float4 f2 = *(const float4*)(row + k0 + 16);
  float4 f3 = *(const float4*)(row + k0 + 20);
  v[0]  = (__bf16)f0.x; v[1]  = (__bf16)f0.y; v[2]  = (__bf16)f0.z; v[3]  = (__bf16)f0.w;
  v[4]  = (__bf16)f1.x; v[5]  = (__bf16)f1.y; v[6]  = (__bf16)f1.z; v[7]  = (__bf16)f1.w;
  v[8]  = (__bf16)f2.x; v[9]  = (__bf16)f2.y; v[10] = (__bf16)f2.z; v[11] = (__bf16)f2.w;
  v[12] = (__bf16)f3.x; v[13] = (__bf16)f3.y; v[14] = (__bf16)f3.z; v[15] = (__bf16)f3.w;
  return v;
}

// ---- device-scope split barrier (monotonic counter; reset by memset each launch)
__device__ inline void grid_signal(unsigned* cnt) {
  __builtin_amdgcn_fence(__ATOMIC_RELEASE, "agent");
  __syncthreads();
  if (threadIdx.x == 0)
    __hip_atomic_fetch_add(cnt, 1u, __ATOMIC_RELAXED, __HIP_MEMORY_SCOPE_AGENT);
}
__device__ inline void grid_wait(unsigned* cnt, unsigned target) {
  if (threadIdx.x == 0) {
    while (__hip_atomic_load(cnt, __ATOMIC_RELAXED, __HIP_MEMORY_SCOPE_AGENT) < target)
      __builtin_amdgcn_s_sleep(1);
  }
  __syncthreads();
  __builtin_amdgcn_fence(__ATOMIC_ACQUIRE, "agent");
}

__global__ void gru_init_h(const float* __restrict__ h0, __bf16* __restrict__ hb) {
  int i = blockIdx.x * blockDim.x + threadIdx.x;
  if (i < GB * GH) hb[i] = (__bf16)h0[i];
}

// One-shot fp32 -> bf16 conversion of x (bandwidth pass: ~96 MB total traffic)
__global__ void gru_x_to_bf16(const float* __restrict__ x, __bf16* __restrict__ xbf,
                              int n4) {
  int stride = gridDim.x * blockDim.x;
  for (int q = blockIdx.x * blockDim.x + threadIdx.x; q < n4; q += stride) {
    float4 f = ((const float4*)x)[q];
    union { __bf16 h[4]; uint2 u; } pk;
    pk.h[0] = (__bf16)f.x; pk.h[1] = (__bf16)f.y;
    pk.h[2] = (__bf16)f.z; pk.h[3] = (__bf16)f.w;
    ((uint2*)xbf)[q] = pk.u;
  }
}

template <bool XBF16>
__global__ __launch_bounds__(NTHR) void gru_scan(
    const float*  __restrict__ x,       // [B,T,I] fp32 (fallback path)
    const __bf16* __restrict__ xbf,     // [B,T,I] bf16 (fast path, ws)
    const float*  __restrict__ h0,      // [B,H]
    const float*  __restrict__ W_ih,    // [3H,I]
    const float*  __restrict__ W_hh,    // [3H,H]
    const float*  __restrict__ b_ih,    // [3H]
    const float*  __restrict__ b_hh,    // [3H]
    const int*    __restrict__ lengths, // [B]
    float* __restrict__ out,            // [B,T,H]
    float* __restrict__ out_final,      // [B,H]
    __bf16* __restrict__ hbuf,          // [2][B][H] bf16 state (ws)
    unsigned* __restrict__ cnt)         // barrier counter (ws)
{
  extern __shared__ __bf16 smem[];            // 2 * 48*512 bf16 = 96 KB
  __bf16* Wih_s = smem;                       // N-major: [48][512]
  __bf16* Whh_s = smem + 48 * GH;

  const int tid  = threadIdx.x;
  const int lane = tid & 31;
  const int wv   = tid >> 5;                  // wave id 0..3 -> batch tile
  const int jb   = blockIdx.x * 16;           // this block's hidden-column base

  // ---- stage weight slices into LDS (fp32 -> bf16), N-major so B-frags are contiguous
  for (int idx = tid; idx < 48 * (GI / 4); idx += NTHR) {
    int n  = idx >> 7;                        // 0..47 (gate-major rows)
    int kq = idx & 127;                       // float4 index along K
    int grow = (n >> 4) * GH + jb + (n & 15); // global gate row in [0,3H)
    float4 a = ((const float4*)(W_ih + (size_t)grow * GI))[kq];
    float4 b = ((const float4*)(W_hh + (size_t)grow * GH))[kq];
    int o = n * GI + kq * 4;
    Wih_s[o + 0] = (__bf16)a.x; Wih_s[o + 1] = (__bf16)a.y;
    Wih_s[o + 2] = (__bf16)a.z; Wih_s[o + 3] = (__bf16)a.w;
    Whh_s[o + 0] = (__bf16)b.x; Whh_s[o + 1] = (__bf16)b.y;
    Whh_s[o + 2] = (__bf16)b.z; Whh_s[o + 3] = (__bf16)b.w;
  }

  // ---- per-lane constants
  const int j   = lane & 15;
  const int jg  = jb + j;                     // global hidden column this lane owns
  const int kg0 = (lane < 16) ? 0 : 8;        // A-frag K sub-offset
  const int kb0 = (lane < 16) ? 0 : 8;        // B-frag K sub-offset
  const int mhalf = (lane < 16) ? 0 : 8;

  const float br   = b_ih[jg]          + b_hh[jg];
  const float bz   = b_ih[GH + jg]     + b_hh[GH + jg];
  const float bin_ = b_ih[2 * GH + jg];
  const float bhn  = b_hh[2 * GH + jg];

  // lane permanently owns 8 (batch,row) cells of h
  float hprev[8];
  int   len[8], brow[8];
#pragma unroll
  for (int i = 0; i < 8; ++i) {
    int b   = 16 * wv + mhalf + i;
    brow[i] = b;
    hprev[i] = h0[(size_t)b * GH + jg];
    len[i]   = lengths[b];
  }

  __syncthreads();

  const int m = 16 * wv + j;                  // A-matrix row this lane feeds
  const __bf16* WsR_i = Wih_s + (size_t)(0  + j) * GI + kb0;
  const __bf16* WsZ_i = Wih_s + (size_t)(16 + j) * GI + kb0;
  const __bf16* WsN_i = Wih_s + (size_t)(32 + j) * GI + kb0;
  const __bf16* WsR_h = Whh_s + (size_t)(0  + j) * GH + kb0;
  const __bf16* WsZ_h = Whh_s + (size_t)(16 + j) * GH + kb0;
  const __bf16* WsN_h = Whh_s + (size_t)(32 + j) * GH + kb0;

  for (int t = 0; t < GT; ++t) {
    v8f accR = {}, accZ = {}, accNi = {}, accNh = {};

    // ---- input projection for step t (independent of h -> overlaps barrier wait)
    if (XBF16) {
      const __bf16* xr = xbf + ((size_t)m * GT + t) * GI;
#pragma unroll 4
      for (int kc = 0; kc < 16; ++kc) {
        v16bf fa = ldfrag_a(xr, kc * 32 + kg0);
        accR  = wmma_bf16(fa, ldfrag_b(WsR_i + kc * 32), accR);
        accZ  = wmma_bf16(fa, ldfrag_b(WsZ_i + kc * 32), accZ);
        accNi = wmma_bf16(fa, ldfrag_b(WsN_i + kc * 32), accNi);
      }
      if (t + 1 < GT) __builtin_prefetch(xr + GI, 0, 0);
    } else {
      const float* xr = x + ((size_t)m * GT + t) * GI;
#pragma unroll 4
      for (int kc = 0; kc < 16; ++kc) {
        v16bf fa = ldfrag_a_f32(xr, kc * 32 + kg0);
        accR  = wmma_bf16(fa, ldfrag_b(WsR_i + kc * 32), accR);
        accZ  = wmma_bf16(fa, ldfrag_b(WsZ_i + kc * 32), accZ);
        accNi = wmma_bf16(fa, ldfrag_b(WsN_i + kc * 32), accNi);
      }
      if (t + 1 < GT) __builtin_prefetch(xr + GI, 0, 0);
    }

    // ---- wait until every block published h for step t
    grid_wait(cnt, (unsigned)(NBLK * t));

    // ---- recurrent projection: h_t @ W_hh^T
    const __bf16* hrow = hbuf + (size_t)(t & 1) * GB * GH + (size_t)m * GH;
#pragma unroll 4
    for (int kc = 0; kc < 16; ++kc) {
      v16bf fa = ldfrag_a(hrow, kc * 32 + kg0);
      accR  = wmma_bf16(fa, ldfrag_b(WsR_h + kc * 32), accR);
      accZ  = wmma_bf16(fa, ldfrag_b(WsZ_h + kc * 32), accZ);
      accNh = wmma_bf16(fa, ldfrag_b(WsN_h + kc * 32), accNh);
    }

    // ---- gates + state update (C-layout: vgpr i -> row mhalf+i, col = lane%16)
    __bf16* hnext = hbuf + (size_t)((t + 1) & 1) * GB * GH;
#pragma unroll
    for (int i = 0; i < 8; ++i) {
      float r  = 1.0f / (1.0f + __expf(-(accR[i] + br)));
      float z  = 1.0f / (1.0f + __expf(-(accZ[i] + bz)));
      float nn = tanhf(accNi[i] + bin_ + r * (accNh[i] + bhn));
      float hn = (1.0f - z) * nn + z * hprev[i];
      bool  valid = (t < len[i]);
      float hk = valid ? hn : hprev[i];
      out[((size_t)brow[i] * GT + t) * GH + jg] = valid ? hn : 0.0f;
      hprev[i] = hk;
      hnext[(size_t)brow[i] * GH + jg] = (__bf16)hk;
    }

    // ---- publish h_{t+1}
    grid_signal(cnt);
  }

  // ---- final state
#pragma unroll
  for (int i = 0; i < 8; ++i)
    out_final[(size_t)brow[i] * GH + jg] = hprev[i];
}

extern "C" void kernel_launch(void* const* d_in, const int* in_sizes, int n_in,
                              void* d_out, int out_size, void* d_ws, size_t ws_size,
                              hipStream_t stream) {
  const float* x       = (const float*)d_in[0];
  const float* h0      = (const float*)d_in[1];
  const float* W_ih    = (const float*)d_in[2];
  const float* W_hh    = (const float*)d_in[3];
  const float* b_ih    = (const float*)d_in[4];
  const float* b_hh    = (const float*)d_in[5];
  const int*   lengths = (const int*)d_in[6];

  float* out       = (float*)d_out;
  float* out_final = out + (size_t)GB * GT * GH;

  const size_t XBF_BYTES  = (size_t)GB * GT * GI * sizeof(__bf16); // 32 MB
  const size_t HBUF_BYTES = (size_t)2 * GB * GH * sizeof(__bf16);  // 128 KB
  const bool   use_xbf    = (ws_size >= XBF_BYTES + HBUF_BYTES + 256);

  __bf16*   xbf  = (__bf16*)d_ws;
  __bf16*   hbuf = (__bf16*)((char*)d_ws + (use_xbf ? XBF_BYTES : 0));
  unsigned* cnt  = (unsigned*)((char*)hbuf + HBUF_BYTES);

  hipMemsetAsync(cnt, 0, sizeof(unsigned), stream);
  gru_init_h<<<(GB * GH + 255) / 256, 256, 0, stream>>>(h0, hbuf);

  if (use_xbf) {
    const int n4 = GB * GT * GI / 4;
    gru_x_to_bf16<<<2048, 256, 0, stream>>>(x, xbf, n4);
    gru_scan<true><<<NBLK, NTHR, 96 * 1024, stream>>>(
        x, xbf, h0, W_ih, W_hh, b_ih, b_hh, lengths, out, out_final, hbuf, cnt);
  } else {
    gru_scan<false><<<NBLK, NTHR, 96 * 1024, stream>>>(
        x, xbf, h0, W_ih, W_hh, b_ih, b_hh, lengths, out, out_final, hbuf, cnt);
  }
}
